// CrossGraphAttentionModel_180388626955
// MI455X (gfx1250) — compile-verified, matrix operand order checked by reading, not run on hardware
//
#include <hip/hip_runtime.h>

#define HID 64
#define HEADS 4
#define HD 16
#define NMOL 2048
#define NPROT 4096
#define EMOL 32768
#define EPROT 131072
#define BATCH 32

typedef __attribute__((ext_vector_type(16))) _Float16 v16h;
typedef __attribute__((ext_vector_type(8)))  _Float16 v8h;
typedef __attribute__((ext_vector_type(8)))  float    v8f;

// ---- WMMA operand layout helpers (ISA 7.12.2, wave32) ---------------------
// A/B 16x32 f16 operand: element [i] of lane l maps to
//   other = l&15 (row for A / col for B), k = 8*(l>>4) + (i&7) + ((i>=8)?16:0)
// C/D 16x16 f32: element [j] of lane l -> row = j + 8*(l>>4), col = l&15.

// Gather-load an operand tile from an f32 matrix (used in small GEMMs).
__device__ __forceinline__ v16h ld_tile(const float* base, int k0, long sK, long sO, int kmax) {
  const int lane  = threadIdx.x & 31;
  const int other = lane & 15;
  const int kb    = (lane >> 4) << 3;
  v16h r;
#pragma unroll
  for (int i = 0; i < 16; ++i) {
    const int k = k0 + kb + (i & 7) + ((i >> 3) << 4);
    float v = (k < kmax) ? base[(long)k * sK + (long)other * sO] : 0.0f;
    r[i] = (_Float16)v;
  }
  return r;
}

// Vector-load 8 f16 (16B) into lower half of an operand; upper K-half = 0.
__device__ __forceinline__ v16h ld_half16(const _Float16* p) {
  v8h lo = *(const v8h*)p;
  v16h r = {};
#pragma unroll
  for (int i = 0; i < 8; ++i) r[i] = lo[i];
  return r;
}

// Vector-load 16 f16 (32B) as a full operand.
__device__ __forceinline__ v16h ld_full16(const _Float16* p) {
  v8h lo = *(const v8h*)p;
  v8h hi = *(const v8h*)(p + 8);
  v16h r;
#pragma unroll
  for (int i = 0; i < 8; ++i) { r[i] = lo[i]; r[i + 8] = hi[i]; }
  return r;
}

__device__ __forceinline__ v8f wmma16(v16h a, v16h b, v8f c) {
  return __builtin_amdgcn_wmma_f32_16x16x32_f16(false, a, false, b, (short)0, c, false, false);
}

// ---------------- small-K linear: Y[N,64] = X[N,K] @ W[K,64] + b ----------
__global__ void k_lin_small(const float* __restrict__ X, const float* __restrict__ W,
                            const float* __restrict__ b, float* __restrict__ Y,
                            int N, int K) {
  long idx = (long)blockIdx.x * blockDim.x + threadIdx.x;
  if (idx >= (long)N * HID) return;
  int r = (int)(idx >> 6), c = (int)(idx & 63);
  float s = b[c];
  for (int j = 0; j < K; ++j) s += X[(long)r * K + j] * W[j * HID + c];
  Y[idx] = s;
}

__global__ void k_copy(const float* __restrict__ src, float* __restrict__ dst, long n) {
  long i = (long)blockIdx.x * blockDim.x + threadIdx.x;
  if (i < n) dst[i] = src[i];
}

__global__ void k_zero(float* __restrict__ p, long n) {
  long i = (long)blockIdx.x * blockDim.x + threadIdx.x;
  if (i < n) p[i] = 0.0f;
}

// ---------------- GINE message + aggregation: h[dst] += relu(x[src]+e) ----
__global__ void k_gine_msg(const float* __restrict__ x, const float* __restrict__ ea,
                           const int* __restrict__ ei, float* __restrict__ h, int E) {
  long idx = (long)blockIdx.x * blockDim.x + threadIdx.x;
  if (idx >= (long)E * HID) return;
  int e = (int)(idx >> 6), c = (int)(idx & 63);
  int s = ei[e], d = ei[E + e];
  float v = x[(long)s * HID + c] + ea[(long)e * HID + c];
  v = v > 0.0f ? v : 0.0f;
  atomicAdd(&h[(long)d * HID + c], v);
}

// ---------------- GINE inner MLP: Y = relu(relu(H@W1+b1)@W2+b2), 16 rows/wave
__global__ void __launch_bounds__(32) k_mlp64(const float* __restrict__ Hin,
                                              const float* __restrict__ W1, const float* __restrict__ b1,
                                              const float* __restrict__ W2, const float* __restrict__ b2,
                                              float* __restrict__ Y) {
  __shared__ float t[16][HID + 4];
  const int row0 = blockIdx.x * 16;
  const int lane = threadIdx.x;
  const int col  = lane & 15, hb = lane >> 4;
  v16h a0 = ld_tile(Hin + (long)row0 * HID, 0, 1, HID, HID);
  v16h a1 = ld_tile(Hin + (long)row0 * HID, 32, 1, HID, HID);
#pragma unroll
  for (int nt = 0; nt < 4; ++nt) {
    v8f acc = {};
    acc = wmma16(a0, ld_tile(W1 + nt * 16, 0, HID, 1, HID), acc);
    acc = wmma16(a1, ld_tile(W1 + nt * 16, 32, HID, 1, HID), acc);
#pragma unroll
    for (int j = 0; j < 8; ++j) {
      float v = acc[j] + b1[nt * 16 + col];
      t[j + hb * 8][nt * 16 + col] = v > 0.0f ? v : 0.0f;
    }
  }
  __syncthreads();
  v16h c0 = ld_tile(&t[0][0], 0, 1, HID + 4, HID);
  v16h c1 = ld_tile(&t[0][0], 32, 1, HID + 4, HID);
#pragma unroll
  for (int nt = 0; nt < 4; ++nt) {
    v8f acc = {};
    acc = wmma16(c0, ld_tile(W2 + nt * 16, 0, HID, 1, HID), acc);
    acc = wmma16(c1, ld_tile(W2 + nt * 16, 32, HID, 1, HID), acc);
#pragma unroll
    for (int j = 0; j < 8; ++j) {
      float v = acc[j] + b2[nt * 16 + col];
      Y[(long)(row0 + j + hb * 8) * HID + nt * 16 + col] = v > 0.0f ? v : 0.0f;
    }
  }
}

// ---- Q/K projection: Y = X@W + b, scaled, packed f16 in scores-operand order
// Qp/Kp layout: [head][node_tile N/16][lane 32][8 f16]
//   lane = (node&15) + 16*(d>>3), slot = d&7   (d = head-dim index 0..15)
__global__ void __launch_bounds__(32) k_lin64_qk(const float* __restrict__ X,
                                                 const float* __restrict__ W, const float* __restrict__ b,
                                                 _Float16* __restrict__ out, int N, float scale) {
  const int row0 = blockIdx.x * 16;
  const int lane = threadIdx.x;
  const int c = lane & 15, hb = lane >> 4;
  v16h a0 = ld_tile(X + (long)row0 * HID, 0, 1, HID, HID);
  v16h a1 = ld_tile(X + (long)row0 * HID, 32, 1, HID, HID);
#pragma unroll
  for (int nt = 0; nt < 4; ++nt) {                 // nt == head
    v8f acc = {};
    acc = wmma16(a0, ld_tile(W + nt * 16, 0, HID, 1, HID), acc);
    acc = wmma16(a1, ld_tile(W + nt * 16, 32, HID, 1, HID), acc);
    const long tbase = ((long)nt * (N >> 4) + blockIdx.x) * 32;
    const int plane_hi = 16 * (c >> 3);
#pragma unroll
    for (int j = 0; j < 8; ++j) {
      float v = (acc[j] + b[nt * 16 + c]) * scale;
      out[(tbase + (j + hb * 8) + plane_hi) * 8 + (c & 7)] = (_Float16)v;
    }
  }
}

// ---- V projection: packed f16 in B-operand order for the AV GEMM.
// Vp layout: [head][k_chunk N/32][lane 32][16 f16]
//   lane = d + 16*((kc>>3)&1), slot = (kc&7) + 8*(kc>=16)   (kc = key node & 31)
__global__ void __launch_bounds__(32) k_lin64_v(const float* __restrict__ X,
                                                const float* __restrict__ W, const float* __restrict__ b,
                                                _Float16* __restrict__ out, int N) {
  const int row0 = blockIdx.x * 16;
  const int lane = threadIdx.x;
  const int c = lane & 15, hb = lane >> 4;
  v16h a0 = ld_tile(X + (long)row0 * HID, 0, 1, HID, HID);
  v16h a1 = ld_tile(X + (long)row0 * HID, 32, 1, HID, HID);
#pragma unroll
  for (int nt = 0; nt < 4; ++nt) {                 // nt == head
    v8f acc = {};
    acc = wmma16(a0, ld_tile(W + nt * 16, 0, HID, 1, HID), acc);
    acc = wmma16(a1, ld_tile(W + nt * 16, 32, HID, 1, HID), acc);
#pragma unroll
    for (int j = 0; j < 8; ++j) {
      int row = row0 + j + hb * 8;
      int chunk = row >> 5, kc = row & 31;
      int lv = c + 16 * ((kc >> 3) & 1);
      int sl = (kc & 7) + 8 * (kc >= 16);
      out[(((long)nt * (N >> 5) + chunk) * 32 + lv) * 16 + sl] =
          (_Float16)(acc[j] + b[nt * 16 + c]);
    }
  }
}

// ---------------- S[h,q,k] = Qp . Kp (scale pre-folded into Qp) ------------
__global__ void __launch_bounds__(32) k_scores(const _Float16* __restrict__ Qp,
                                               const _Float16* __restrict__ Kp,
                                               float* __restrict__ S, int Nq, int Nk) {
  const int kt = blockIdx.x, qt = blockIdx.y, h = blockIdx.z;
  const int lane = threadIdx.x, col = lane & 15, hb = lane >> 4;
  v16h a  = ld_half16(Qp + (((long)h * (Nq >> 4) + qt) * 32 + lane) * 8);
  v16h bb = ld_half16(Kp + (((long)h * (Nk >> 4) + kt) * 32 + lane) * 8);
  v8f acc = {};
  acc = wmma16(a, bb, acc);
  float* Sp = S + ((long)h * Nq + qt * 16) * Nk + kt * 16;
#pragma unroll
  for (int j = 0; j < 8; ++j)
    Sp[(long)(j + hb * 8) * Nk + col] = acc[j];
}

// ---- row softmax over Nk; final pass emits normalized f16 probs packed in
// A-operand order: Pp[h][q_tile][k_chunk Nk/32][lane 32][16 f16]
__global__ void k_softmax_pack(float* __restrict__ S, _Float16* __restrict__ Pp,
                               int Nq, int Nk) {
  const long row = blockIdx.x;                 // h*Nq + q
  const int h = (int)(row / Nq), q = (int)(row % Nq);
  float* p = S + row * (long)Nk;
  __shared__ float red[256];
  const int tid = threadIdx.x;
  float m = -1e30f;
  for (int i = tid; i < Nk; i += 256) m = fmaxf(m, p[i]);
  red[tid] = m; __syncthreads();
  for (int s = 128; s > 0; s >>= 1) { if (tid < s) red[tid] = fmaxf(red[tid], red[tid + s]); __syncthreads(); }
  m = red[0]; __syncthreads();
  float sum = 0.0f;
  for (int i = tid; i < Nk; i += 256) { float e = __expf(p[i] - m); p[i] = e; sum += e; }
  red[tid] = sum; __syncthreads();
  for (int s = 128; s > 0; s >>= 1) { if (tid < s) red[tid] += red[tid + s]; __syncthreads(); }
  const float inv = 1.0f / red[0];
  _Float16* base = Pp + ((long)h * (Nq >> 4) + (q >> 4)) * (long)(Nk >> 5) * 512;
  const int qm = q & 15;
  for (int k = tid; k < Nk; k += 256) {
    int chunk = k >> 5, kc = k & 31;
    int lv = qm + 16 * ((kc >> 3) & 1);
    int sl = (kc & 7) + 8 * (kc >= 16);
    base[((long)chunk * 32 + lv) * 16 + sl] = (_Float16)(p[k] * inv);
  }
}

// ---------------- H[q, h*16+n] = Xres + sum_k P[h,q,k]*V[k, h*16+n] --------
__global__ void __launch_bounds__(32) k_av(const _Float16* __restrict__ Pp,
                                           const _Float16* __restrict__ Vp,
                                           const float* __restrict__ Xres, float* __restrict__ H,
                                           int Nq, int Nk) {
  const int qt = blockIdx.x, h = blockIdx.y;
  const int lane = threadIdx.x, col = lane & 15, hb = lane >> 4;
  const _Float16* Pbase = Pp + ((long)h * (Nq >> 4) + qt) * (long)(Nk >> 5) * 512;
  const _Float16* Vbase = Vp + (long)h * (Nk >> 5) * 512;
  v8f acc = {};
  const int nchunk = Nk >> 5;
  for (int c = 0; c < nchunk; ++c) {
    if (c + 1 < nchunk) {
      __builtin_prefetch(Pbase + ((long)(c + 1) * 32 + lane) * 16, 0, 0);
      __builtin_prefetch(Vbase + ((long)(c + 1) * 32 + lane) * 16, 0, 0);
    }
    v16h a  = ld_full16(Pbase + ((long)c * 32 + lane) * 16);
    v16h bb = ld_full16(Vbase + ((long)c * 32 + lane) * 16);
    acc = wmma16(a, bb, acc);
  }
#pragma unroll
  for (int j = 0; j < 8; ++j) {
    long r = (long)qt * 16 + j + hb * 8;
    H[r * HID + h * HD + col] = Xres[r * HID + h * HD + col] + acc[j];
  }
}

// ---------------- mean pool ------------------------------------------------
__global__ void k_pool_acc(const float* __restrict__ H, const int* __restrict__ batch,
                           float* __restrict__ z, float* __restrict__ cnt, int N, int off) {
  long idx = (long)blockIdx.x * blockDim.x + threadIdx.x;
  if (idx >= (long)N * HID) return;
  int i = (int)(idx >> 6), c = (int)(idx & 63);
  int b = batch[i];
  atomicAdd(&z[(long)b * 2 * HID + off + c], H[idx]);
  if (c == 0) atomicAdd(&cnt[b], 1.0f);
}

__global__ void k_pool_div(float* __restrict__ z, const float* __restrict__ cm,
                           const float* __restrict__ cp) {
  int idx = blockIdx.x * blockDim.x + threadIdx.x;
  if (idx >= BATCH * 2 * HID) return;
  int b = idx >> 7, c = idx & 127;
  float cnt = (c < HID) ? cm[b] : cp[b];
  z[idx] /= fmaxf(cnt, 1.0f);
}

// ---------------- head: sigmoid(relu(z@W1+b1)@W2+b2) -----------------------
__global__ void k_head(const float* __restrict__ z, const float* __restrict__ W1,
                       const float* __restrict__ b1, const float* __restrict__ W2,
                       const float* __restrict__ b2, float* __restrict__ out) {
  const int b = blockIdx.x, c = threadIdx.x;
  __shared__ float red[64];
  float s = b1[c];
  for (int i = 0; i < 2 * HID; ++i) s += z[b * 2 * HID + i] * W1[i * HID + c];
  s = fmaxf(s, 0.0f);
  red[c] = s * W2[c];
  __syncthreads();
  for (int st = 32; st > 0; st >>= 1) { if (c < st) red[c] += red[c + st]; __syncthreads(); }
  if (c == 0) out[b] = 1.0f / (1.0f + __expf(-(red[0] + b2[0])));
}

extern "C" void kernel_launch(void* const* d_in, const int* in_sizes, int n_in,
                              void* d_out, int out_size, void* d_ws, size_t ws_size,
                              hipStream_t stream) {
  (void)in_sizes; (void)n_in; (void)out_size; (void)ws_size;

  const float* mol_x      = (const float*)d_in[0];
  const float* prot_x     = (const float*)d_in[1];
  const float* mol_eattr  = (const float*)d_in[2];
  const float* prot_eattr = (const float*)d_in[3];
  const int*   mol_ei     = (const int*)d_in[4];
  const int*   prot_ei    = (const int*)d_in[5];
  const int*   mol_batch  = (const int*)d_in[6];
  const int*   prot_batch = (const int*)d_in[7];
  const float* nlmW = (const float*)d_in[8],  *nlmb = (const float*)d_in[9];
  const float* nlpW = (const float*)d_in[10], *nlpb = (const float*)d_in[11];
  const float* elmW = (const float*)d_in[12], *elmb = (const float*)d_in[13];
  const float* elpW = (const float*)d_in[14], *elpb = (const float*)d_in[15];
  const float* mcW1 = (const float*)d_in[16], *mcb1 = (const float*)d_in[17];
  const float* mcW2 = (const float*)d_in[18], *mcb2 = (const float*)d_in[19];
  const float* pcW1 = (const float*)d_in[20], *pcb1 = (const float*)d_in[21];
  const float* pcW2 = (const float*)d_in[22], *pcb2 = (const float*)d_in[23];
  const float* ampW = (const float*)d_in[24], *ampb = (const float*)d_in[25];
  const float* apmW = (const float*)d_in[26], *apmb = (const float*)d_in[27];
  const float* fc1W = (const float*)d_in[28], *fc1b = (const float*)d_in[29];
  const float* fc2W = (const float*)d_in[30], *fc2b = (const float*)d_in[31];

  char* wsb = (char*)d_ws;
  size_t off = 0;
  auto alloc = [&](size_t bytes) {
    off = (off + 255) & ~(size_t)255;
    void* p = wsb + off;
    off += bytes;
    return p;
  };
  float* xm = (float*)alloc((size_t)NMOL * HID * 4);
  float* xp = (float*)alloc((size_t)NPROT * HID * 4);
  float* em = (float*)alloc((size_t)EMOL * HID * 4);
  float* ep = (float*)alloc((size_t)EPROT * HID * 4);
  float* tm = (float*)alloc((size_t)NMOL * HID * 4);
  float* tp = (float*)alloc((size_t)NPROT * HID * 4);
  float* S  = (float*)alloc((size_t)HEADS * NMOL * NPROT * 4);
  float* Hm = (float*)alloc((size_t)NMOL * HID * 4);
  float* Hp = (float*)alloc((size_t)NPROT * HID * 4);
  float* z  = (float*)alloc((size_t)(BATCH * 2 * HID + 2 * BATCH) * 4);
  float* cm = z + BATCH * 2 * HID;
  float* cp = cm + BATCH;
  _Float16* Qp = (_Float16*)alloc((size_t)NPROT * HID * 2);   // [h][tile][32][8]
  _Float16* Kp = (_Float16*)alloc((size_t)NPROT * HID * 2);
  _Float16* Vp = (_Float16*)alloc((size_t)NPROT * HID * 2);   // [h][chunk][32][16]
  _Float16* Pp = (_Float16*)alloc((size_t)HEADS * NMOL * NPROT * 2);

  const int T = 256;
  // input projections
  k_lin_small<<<(NMOL * HID + T - 1) / T, T, 0, stream>>>(mol_x, nlmW, nlmb, xm, NMOL, 11);
  k_lin_small<<<(NPROT * HID + T - 1) / T, T, 0, stream>>>(prot_x, nlpW, nlpb, xp, NPROT, 15);
  k_lin_small<<<(EMOL * HID + T - 1) / T, T, 0, stream>>>(mol_eattr, elmW, elmb, em, EMOL, 10);
  k_lin_small<<<(EPROT * HID + T - 1) / T, T, 0, stream>>>(prot_eattr, elpW, elpb, ep, EPROT, 10);

  // 3 GINE layers per graph
  for (int l = 0; l < 3; ++l) {
    k_copy<<<(NMOL * HID + T - 1) / T, T, 0, stream>>>(xm, tm, (long)NMOL * HID);
    k_gine_msg<<<((long)EMOL * HID + T - 1) / T, T, 0, stream>>>(xm, em, mol_ei, tm, EMOL);
    k_mlp64<<<NMOL / 16, 32, 0, stream>>>(tm, mcW1 + l * 4096, mcb1 + l * 64,
                                          mcW2 + l * 4096, mcb2 + l * 64, xm);
  }
  for (int l = 0; l < 3; ++l) {
    k_copy<<<(NPROT * HID + T - 1) / T, T, 0, stream>>>(xp, tp, (long)NPROT * HID);
    k_gine_msg<<<((long)EPROT * HID + T - 1) / T, T, 0, stream>>>(xp, ep, prot_ei, tp, EPROT);
    k_mlp64<<<NPROT / 16, 32, 0, stream>>>(tp, pcW1 + l * 4096, pcb1 + l * 64,
                                           pcW2 + l * 4096, pcb2 + l * 64, xp);
  }

  const float qscale = 0.25f;  // 1/sqrt(HEAD_DIM)

  // cross attention: mol queries over prot keys
  k_lin64_qk<<<NMOL / 16, 32, 0, stream>>>(xm, ampW, ampb, Qp, NMOL, qscale);
  k_lin64_qk<<<NPROT / 16, 32, 0, stream>>>(xp, ampW + 4096, ampb + 64, Kp, NPROT, 1.0f);
  k_lin64_v<<<NPROT / 16, 32, 0, stream>>>(xp, ampW + 8192, ampb + 128, Vp, NPROT);
  {
    dim3 gs(NPROT / 16, NMOL / 16, HEADS);
    k_scores<<<gs, 32, 0, stream>>>(Qp, Kp, S, NMOL, NPROT);
    k_softmax_pack<<<HEADS * NMOL, 256, 0, stream>>>(S, Pp, NMOL, NPROT);
    dim3 ga(NMOL / 16, HEADS);
    k_av<<<ga, 32, 0, stream>>>(Pp, Vp, xm, Hm, NMOL, NPROT);
  }

  // cross attention: prot queries over mol keys
  k_lin64_qk<<<NPROT / 16, 32, 0, stream>>>(xp, apmW, apmb, Qp, NPROT, qscale);
  k_lin64_qk<<<NMOL / 16, 32, 0, stream>>>(xm, apmW + 4096, apmb + 64, Kp, NMOL, 1.0f);
  k_lin64_v<<<NMOL / 16, 32, 0, stream>>>(xm, apmW + 8192, apmb + 128, Vp, NMOL);
  {
    dim3 gs(NMOL / 16, NPROT / 16, HEADS);
    k_scores<<<gs, 32, 0, stream>>>(Qp, Kp, S, NPROT, NMOL);
    k_softmax_pack<<<HEADS * NPROT, 256, 0, stream>>>(S, Pp, NPROT, NMOL);
    dim3 ga(NPROT / 16, HEADS);
    k_av<<<ga, 32, 0, stream>>>(Pp, Vp, xp, Hp, NPROT, NMOL);
  }

  // mean pool + head
  k_zero<<<(BATCH * 2 * HID + 2 * BATCH + T - 1) / T, T, 0, stream>>>(z, BATCH * 2 * HID + 2 * BATCH);
  k_pool_acc<<<((long)NMOL * HID + T - 1) / T, T, 0, stream>>>(Hm, mol_batch, z, cm, NMOL, 0);
  k_pool_acc<<<((long)NPROT * HID + T - 1) / T, T, 0, stream>>>(Hp, prot_batch, z, cp, NPROT, HID);
  k_pool_div<<<(BATCH * 2 * HID + T - 1) / T, T, 0, stream>>>(z, cm, cp);
  k_head<<<BATCH, 64, 0, stream>>>(z, fc1W, fc1b, fc2W, fc2b, (float*)d_out);
}
